// NsaAttention_19799799234578
// MI455X (gfx1250) — matrix operand order repaired and checked by reading, not compiled, not run on hardware
//
#include <hip/hip_runtime.h>
#include <cstdint>

typedef __attribute__((ext_vector_type(16))) _Float16 v16h;
typedef __attribute__((ext_vector_type(8)))  _Float16 v8h;
typedef __attribute__((ext_vector_type(8)))  float    v8f;

#define S_    2048
#define H_    16
#define D_    128
#define CP_   128      // padded compressed-block count
#define CN_   127      // real compressed-block count
#define NSEL_ 32
#define WIN_  512
#define SCALE_ 0.08838834764831845f
#define NEGF_  (-1.0e30f)

static __device__ __forceinline__ v8f vzero8() {
  v8f z = {0.f,0.f,0.f,0.f,0.f,0.f,0.f,0.f};
  return z;
}

static __device__ __forceinline__ v8f wmma16(v16h a, v16h b, v8f c) {
  // D = A(16xK32 f16) * B(K32x16 f16) + C(f32)
  return __builtin_amdgcn_wmma_f32_16x16x32_f16(false, a, false, b, (short)0, c,
                                                false, false);
}

// A-fragment loader (also used for Vt rows): lane l holds M=l&15;
// halves 0..15 = K base+(hi?16:0)+0..15 (contiguous)
static __device__ __forceinline__ v16h ld_afrag(const _Float16* rp, int base, int hi) {
  const int o = base + hi * 16;
  v8h lo = *(const v8h*)(rp + o);
  v8h hh = *(const v8h*)(rp + o + 8);
  v16h f;
#pragma unroll
  for (int i = 0; i < 8; ++i) { f[i] = lo[i]; f[8 + i] = hh[i]; }
  return f;
}

// ---------------- precompute kernels ----------------

__global__ void nsa_prep_convert(const float* __restrict__ k, const float* __restrict__ v,
                                 _Float16* __restrict__ Kf, _Float16* __restrict__ Vt) {
  int idx = blockIdx.x * blockDim.x + threadIdx.x;   // (h*S + s)*D + d
  if (idx >= H_ * S_ * D_) return;
  int d = idx & (D_ - 1);
  int s = (idx >> 7) & (S_ - 1);
  int h = idx >> 18;
  float kv = k[((size_t)s * H_ + h) * D_ + d];
  float vv = v[((size_t)s * H_ + h) * D_ + d];
  Kf[idx] = (_Float16)kv;                            // [h][s][d]
  Vt[((size_t)(h * D_ + d)) * S_ + s] = (_Float16)vv; // [h][d][s]
}

__global__ void nsa_prep_cmp(const float* __restrict__ k, const float* __restrict__ v,
                             _Float16* __restrict__ cK, _Float16* __restrict__ cVt) {
  int idx = blockIdx.x * blockDim.x + threadIdx.x;   // (h*CP + c)*D + d
  if (idx >= H_ * CP_ * D_) return;
  int d = idx & (D_ - 1);
  int c = (idx >> 7) & (CP_ - 1);
  int h = idx >> 14;
  float sk = 0.f, sv = 0.f;
  if (c < CN_) {
    int s0 = c * 16;
#pragma unroll 4
    for (int i = 0; i < 32; ++i) {
      sk += k[((size_t)(s0 + i) * H_ + h) * D_ + d];
      sv += v[((size_t)(s0 + i) * H_ + h) * D_ + d];
    }
    sk *= (1.f / 32.f);
    sv *= (1.f / 32.f);
  }
  cK[idx] = (_Float16)sk;                              // [h][c][d]
  cVt[((size_t)(h * D_ + d)) * CP_ + c] = (_Float16)sv; // [h][d][c]
}

// ---------------- main fused NSA kernel: one wave per (head, 16-query tile) ----

__global__ __launch_bounds__(32) void nsa_main_19799799234578(
    const float* __restrict__ q, const float* __restrict__ wg, const float* __restrict__ bg,
    const _Float16* __restrict__ Kf, const _Float16* __restrict__ Vt,
    const _Float16* __restrict__ cK, const _Float16* __restrict__ cVt,
    float* __restrict__ out) {
  const int gid  = blockIdx.x;
  const int h    = gid >> 7;        // 16 heads
  const int qt   = gid & 127;       // 128 query tiles
  const int qb   = qt * 16;
  const int lane = threadIdx.x;
  const int qi   = lane & 15;
  const int hi   = lane >> 4;
  const int t    = qb + qi;         // this lane's query position

  const _Float16* Kh  = Kf  + (size_t)h * S_ * D_;
  const _Float16* Vh  = Vt  + (size_t)h * D_ * S_;
  const _Float16* cKh = cK  + (size_t)h * CP_ * D_;
  const _Float16* cVh = cVt + (size_t)h * D_ * CP_;
  const float*    qrow = q + ((size_t)t * H_ + h) * D_;

  // ---- Q as B-fragments: lane=query col; halves = 16 contiguous dqk values
  v16h qf[4];
#pragma unroll
  for (int u = 0; u < 4; ++u) {
    const float* p = qrow + u * 32 + hi * 16;
#pragma unroll
    for (int i = 0; i < 16; ++i) qf[u][i] = (_Float16)p[i];
  }

  // S^T tile: rows = 16 keys at kt.., cols = queries
  auto qk_tile = [&](const _Float16* kmat, int kt) -> v8f {
    const _Float16* rp = kmat + (size_t)(kt + qi) * D_;
    v8f s = vzero8();
#pragma unroll
    for (int u = 0; u < 4; ++u) s = wmma16(ld_afrag(rp, u * 32, hi), qf[u], s);
    return s;
  };

  // exp'd score tiles (keys kb..kb+31) -> B-fragment of P^T via lane-pair swap
  auto make_bfrag = [&](const v8f& t0, const v8f& t1) -> v16h {
    union { _Float16 hh[2]; unsigned int u; } cv;
    unsigned int p0[4], p1[4], q0[4], q1[4];
#pragma unroll
    for (int i = 0; i < 4; ++i) {
      cv.hh[0] = (_Float16)t0[2*i]; cv.hh[1] = (_Float16)t0[2*i+1]; p0[i] = cv.u;
      cv.hh[0] = (_Float16)t1[2*i]; cv.hh[1] = (_Float16)t1[2*i+1]; p1[i] = cv.u;
    }
#pragma unroll
    for (int i = 0; i < 4; ++i) {
      q0[i] = __shfl_xor(p0[i], 16, 32);
      q1[i] = __shfl_xor(p1[i], 16, 32);
    }
    union { unsigned int w[8]; v16h f; } bb;
#pragma unroll
    for (int i = 0; i < 4; ++i) {
      bb.w[i]     = hi ? q1[i] : p0[i];   // halves 0..7 : keys kb+0..7 | kb+16..23
      bb.w[4 + i] = hi ? p1[i] : q0[i];   // halves 8..15: keys kb+8..15| kb+24..31
    }
    return bb.f;
  };

  // ======== branch 1: compressed attention (materialize all 128 c-scores) ====
  v8f e[8];
#pragma unroll
  for (int ct = 0; ct < 8; ++ct) e[ct] = qk_tile(cKh, ct * 16);

  float mx = NEGF_;
#pragma unroll
  for (int ct = 0; ct < 8; ++ct)
#pragma unroll
    for (int r = 0; r < 8; ++r) {
      int  c   = ct * 16 + r + hi * 8;
      bool val = (c < CN_) && (c * 16 + 31 <= t);
      float sv = val ? e[ct][r] * SCALE_ : NEGF_;
      e[ct][r] = sv;
      mx = fmaxf(mx, sv);
    }
  mx = fmaxf(mx, __shfl_xor(mx, 16, 32));
  float l1 = 0.f;
#pragma unroll
  for (int ct = 0; ct < 8; ++ct)
#pragma unroll
    for (int r = 0; r < 8; ++r) {
      float sv = e[ct][r];
      float ev = (sv > 0.5f * NEGF_) ? expf(sv - mx) : 0.f;
      e[ct][r] = ev;      // unnormalized probs
      l1 += ev;
    }
  l1 += __shfl_xor(l1, 16, 32);

  // ---- selection scores: p_sel[j] = sum of p_cmp[c] for c in {4j-1..4j+3}
  float ps[NSEL_];
#pragma unroll
  for (int j = 0; j < NSEL_; ++j) ps[j] = 0.f;
  if (hi == 0) {
#pragma unroll
    for (int ct = 0; ct < 8; ++ct)
#pragma unroll
      for (int r = 0; r < 8; ++r) {          // c = 16ct + r
        float p = e[ct][r];
        ps[4 * ct + (r >> 2)] += p;
        if ((r & 3) == 3) ps[4 * ct + (r >> 2) + 1] += p;
      }
  } else {
#pragma unroll
    for (int ct = 0; ct < 8; ++ct)
#pragma unroll
      for (int r = 0; r < 8; ++r) {          // c = 16ct + 8 + r
        float p = e[ct][r];
        ps[4 * ct + 2 + (r >> 2)] += p;
        if ((r & 3) == 3) {
          int jj = 4 * ct + 2 + (r >> 2) + 1;
          if (jj < NSEL_) ps[jj] += p;
        }
      }
  }
#pragma unroll
  for (int j = 0; j < NSEL_; ++j) ps[j] += __shfl_xor(ps[j], 16, 32);

  // ---- top-16 of 32 (lowest-index tie-break) via LDS scan
  __shared__ float scb[32][NSEL_ + 1];
  {
    int jcur = t >> 6;
#pragma unroll
    for (int j = 0; j < NSEL_; ++j) {
      float v = (j * 64 <= t) ? ps[j] : -1.0f;
      if (j == jcur) v += 1e6f;
      scb[lane][j] = v;
    }
  }
  unsigned int sel = 0u;
  for (int n = 0; n < 16; ++n) {
    float best = -3.0e38f; int bi = 0;
    for (int j = 0; j < NSEL_; ++j) {
      if ((sel >> j) & 1u) continue;
      float v = scb[lane][j];
      if (v > best) { best = v; bi = j; }
    }
    sel |= (1u << bi);
  }
  unsigned int U = sel;   // union over the 16 queries (uniform after reduce)
#pragma unroll
  for (int o = 1; o < 32; o <<= 1) U |= __shfl_xor(U, o, 32);

  // ---- gates (per-query; replicated on lane pair)
  float g0, g1, g2;
  {
    float a0 = bg[0], a1 = bg[1], a2 = bg[2];
    for (int d = 0; d < D_; ++d) {
      float qv = qrow[d];
      a0 += qv * wg[d * 3 + 0];
      a1 += qv * wg[d * 3 + 1];
      a2 += qv * wg[d * 3 + 2];
    }
    g0 = 1.f / (1.f + expf(-a0));
    g1 = 1.f / (1.f + expf(-a1));
    g2 = 1.f / (1.f + expf(-a2));
  }

  // ---- branch-1 PV: O1^T = cVt * P^T, accumulate into fin
  v8f fin[8];
#pragma unroll
  for (int dt = 0; dt < 8; ++dt) fin[dt] = vzero8();
#pragma unroll
  for (int u = 0; u < 4; ++u) {
    v16h b = make_bfrag(e[2 * u], e[2 * u + 1]);
#pragma unroll
    for (int dt = 0; dt < 8; ++dt) {
      const _Float16* rp = cVh + (size_t)(dt * 16 + qi) * CP_;
      fin[dt] = wmma16(ld_afrag(rp, u * 32, hi), b, fin[dt]);
    }
  }
  {
    float w1 = g0 * (l1 > 0.f ? 1.f / l1 : 0.f);
#pragma unroll
    for (int dt = 0; dt < 8; ++dt)
#pragma unroll
      for (int r = 0; r < 8; ++r) fin[dt][r] *= w1;
  }

  // ======== flash chunk (32 keys) for branches 2 & 3 =========================
  auto chunk = [&](int kb, bool window, unsigned int smask,
                   float& m, float& l, v8f* acc) {
    v8f s0 = qk_tile(Kh, kb);
    v8f s1 = qk_tile(Kh, kb + 16);
    float cm = NEGF_;
#pragma unroll
    for (int r = 0; r < 8; ++r) {
      int k0 = kb + r + hi * 8;
      int k1 = k0 + 16;
      bool v0, v1;
      if (window) {
        v0 = (k0 <= t) && (t - k0 <= WIN_);
        v1 = (k1 <= t) && (t - k1 <= WIN_);
      } else {
        v0 = (k0 <= t) && ((smask >> (k0 >> 6)) & 1u);
        v1 = (k1 <= t) && ((smask >> (k1 >> 6)) & 1u);
      }
      float a = v0 ? s0[r] * SCALE_ : NEGF_;
      float b = v1 ? s1[r] * SCALE_ : NEGF_;
      s0[r] = a; s1[r] = b;
      cm = fmaxf(cm, fmaxf(a, b));
    }
    cm = fmaxf(cm, __shfl_xor(cm, 16, 32));
    float mn  = fmaxf(m, cm);
    float scl = expf(m - mn);
    float ls  = 0.f;
#pragma unroll
    for (int r = 0; r < 8; ++r) {
      float e0 = (s0[r] > 0.5f * NEGF_) ? expf(s0[r] - mn) : 0.f;
      float e1 = (s1[r] > 0.5f * NEGF_) ? expf(s1[r] - mn) : 0.f;
      s0[r] = e0; s1[r] = e1;
      ls += e0 + e1;
    }
    ls += __shfl_xor(ls, 16, 32);
    l = l * scl + ls;
    v16h b = make_bfrag(s0, s1);
#pragma unroll
    for (int dt = 0; dt < 8; ++dt) {
      v8f c;
#pragma unroll
      for (int r = 0; r < 8; ++r) c[r] = acc[dt][r] * scl;
      const _Float16* rp = Vh + (size_t)(dt * 16 + qi) * S_;
      acc[dt] = wmma16(ld_afrag(rp, kb, hi), b, c);
    }
    m = mn;
  };

  const int tmax = qb + 15;

  // ======== branch 2: selected-block attention over union of selected blocks
  {
    float m2 = NEGF_, l2 = 0.f;
    v8f a2[8];
#pragma unroll
    for (int dt = 0; dt < 8; ++dt) a2[dt] = vzero8();
    for (int j = 0; j < NSEL_; ++j) {
      if (!((U >> j) & 1u)) continue;
      int b0 = j * 64;
      if (b0 > tmax) continue;
      chunk(b0, false, sel, m2, l2, a2);
      if (b0 + 32 <= tmax) chunk(b0 + 32, false, sel, m2, l2, a2);
    }
    float w = g1 * (l2 > 0.f ? 1.f / l2 : 0.f);
#pragma unroll
    for (int dt = 0; dt < 8; ++dt)
#pragma unroll
      for (int r = 0; r < 8; ++r) fin[dt][r] += a2[dt][r] * w;
  }

  // ======== branch 3: sliding-window causal attention =======================
  {
    float m3 = NEGF_, l3 = 0.f;
    v8f a3[8];
#pragma unroll
    for (int dt = 0; dt < 8; ++dt) a3[dt] = vzero8();
    int k0 = qb - WIN_;
    if (k0 < 0) k0 = 0;
    k0 &= ~31;
    for (int kb = k0; kb <= tmax; kb += 32) chunk(kb, true, 0u, m3, l3, a3);
    float w = g2 * (l3 > 0.f ? 1.f / l3 : 0.f);
#pragma unroll
    for (int dt = 0; dt < 8; ++dt)
#pragma unroll
      for (int r = 0; r < 8; ++r) fin[dt][r] += a3[dt][r] * w;
  }

  // ---- store: fin is O^T C-layout -> contiguous 8-float runs per tile
  float* orow = out + ((size_t)t * H_ + h) * D_;
#pragma unroll
  for (int dt = 0; dt < 8; ++dt) {
    int d0 = dt * 16 + hi * 8;
    float4 x0 = make_float4(fin[dt][0], fin[dt][1], fin[dt][2], fin[dt][3]);
    float4 x1 = make_float4(fin[dt][4], fin[dt][5], fin[dt][6], fin[dt][7]);
    *(float4*)(orow + d0)     = x0;
    *(float4*)(orow + d0 + 4) = x1;
  }
}

// ---------------- launcher ----------------

extern "C" void kernel_launch(void* const* d_in, const int* in_sizes, int n_in,
                              void* d_out, int out_size, void* d_ws, size_t ws_size,
                              hipStream_t stream) {
  (void)in_sizes; (void)n_in; (void)out_size; (void)ws_size;
  const float* q  = (const float*)d_in[0];
  const float* k  = (const float*)d_in[1];
  const float* v  = (const float*)d_in[2];
  const float* wg = (const float*)d_in[3];
  const float* bg = (const float*)d_in[4];
  float* out = (float*)d_out;

  char* ws = (char*)d_ws;
  const size_t szK = (size_t)H_ * S_ * D_ * sizeof(_Float16);   // 8 MB
  const size_t szC = (size_t)H_ * CP_ * D_ * sizeof(_Float16);  // 512 KB
  _Float16* Kf  = (_Float16*)(ws);
  _Float16* Vt  = (_Float16*)(ws + szK);
  _Float16* cK  = (_Float16*)(ws + 2 * szK);
  _Float16* cVt = (_Float16*)(ws + 2 * szK + szC);

  int n1 = H_ * S_ * D_;
  nsa_prep_convert<<<(n1 + 255) / 256, 256, 0, stream>>>(k, v, Kf, Vt);
  int n2 = H_ * CP_ * D_;
  nsa_prep_cmp<<<(n2 + 255) / 256, 256, 0, stream>>>(k, v, cK, cVt);
  nsa_main_19799799234578<<<128 * H_, 32, 0, stream>>>(q, wg, bg, Kf, Vt, cK, cVt, out);
}
